// Grapher_343597384470
// MI455X (gfx1250) — matrix operand order repaired and checked by reading, not compiled, not run on hardware
//
#include <hip/hip_runtime.h>
#include <hip/hip_bf16.h>
#include <math.h>

typedef __bf16 bf16;
typedef __attribute__((ext_vector_type(16))) __bf16 v16bf;
typedef __attribute__((ext_vector_type(8)))  __bf16 v8bf;
typedef __attribute__((ext_vector_type(8)))  float  v8f;

#define CC    192
#define C2C   384
#define CG    96
#define NSP   3136      // 56*56
#define BB    8
#define MROWS (BB*NSP)  // 25088
#define KNN   9
#define PLD   392       // padded LDS panel row (bf16 elems): stride 784B = 196 dwords (== 4 mod 64 banks)

// ---------------- WMMA helpers ----------------

__device__ __forceinline__ v8f wmma_bf16(v16bf a, v16bf b, v8f c) {
  return __builtin_amdgcn_wmma_f32_16x16x32_bf16(false, a, false, b, (short)0, c, false, false);
}

// A fragment: 16x32 bf16, rows row0..row0+15, k0..k0+31, row-major source (ld elems)
// lane m = lane&15; lanes<16: K={k0..k0+7, k0+16..k0+23}; lanes>=16: +8
__device__ __forceinline__ v16bf load_frag_a(const bf16* base, int row0, int k0, int ld, int lane) {
  int m    = lane & 15;
  int koff = k0 + ((lane & 16) ? 8 : 0);
  const bf16* p = base + (size_t)(row0 + m) * ld + koff;
  union { v16bf v; v8bf h[2]; } u;
  u.h[0] = *(const v8bf*)p;
  u.h[1] = *(const v8bf*)(p + 16);
  return u.v;
}

// B fragment: 32x16 bf16 where B[k][n] = W[col0+n][k0+k]; W row-major (ld elems).
// lane n = lane&15; lanes<16: K={k0..k0+7, k0+16..k0+23}; lanes>=16: +8
__device__ __forceinline__ v16bf load_frag_b(const bf16* base, int col0, int k0, int ld, int lane) {
  int n    = lane & 15;
  int koff = k0 + ((lane & 16) ? 8 : 0);
  const bf16* p = base + (size_t)(col0 + n) * ld + koff;
  union { v16bf v; v8bf h[2]; } u;
  u.h[0] = *(const v8bf*)p;
  u.h[1] = *(const v8bf*)(p + 16);
  return u.v;
}

// branch-free-ish unrolled top-9 insert (registers only; descending sort)
__device__ __forceinline__ void topk_insert(float (&val)[KNN], int (&idx)[KNN], float v, int c) {
  if (v <= val[KNN - 1]) return;
  #pragma unroll
  for (int t = KNN - 1; t >= 1; --t) {
    if (v > val[t - 1])      { val[t] = val[t - 1]; idx[t] = idx[t - 1]; }
    else if (v > val[t])     { val[t] = v;          idx[t] = c;          }
  }
  if (v > val[0]) { val[0] = v; idx[0] = c; }
}

// ---------------- kernels ----------------

__global__ void f2bf_kernel(const float* __restrict__ src, bf16* __restrict__ dst, int n) {
  int i = blockIdx.x * 256 + threadIdx.x;
  if (i < n) dst[i] = (bf16)src[i];
}

// x (B,C,N) fp32 -> xT (B,N,C) bf16; block (32,8), grid (N/32, C/32, B)
__global__ void transpose_kernel(const float* __restrict__ x, bf16* __restrict__ xT) {
  __shared__ float tile[32][33];
  int b  = blockIdx.z;
  int n0 = blockIdx.x * 32, c0 = blockIdx.y * 32;
  const float* xb = x + (size_t)b * CC * NSP;
  #pragma unroll
  for (int i = 0; i < 32; i += 8)
    tile[threadIdx.y + i][threadIdx.x] = xb[(size_t)(c0 + threadIdx.y + i) * NSP + n0 + threadIdx.x];
  __syncthreads();
  bf16* xTb = xT + (size_t)b * NSP * CC;
  #pragma unroll
  for (int i = 0; i < 32; i += 8)
    xTb[(size_t)(n0 + threadIdx.y + i) * CC + c0 + threadIdx.x] = (bf16)tile[threadIdx.x][threadIdx.y + i];
}

// h = BN1(xT @ fc1_w^T + fc1_b)  -> fp32 [M,192]; one 16x16 tile per wave
__global__ void fc1_kernel(const bf16* __restrict__ xT, const bf16* __restrict__ w1,
                           const float* __restrict__ fb, const float* __restrict__ g1,
                           const float* __restrict__ b1, const float* __restrict__ m1,
                           const float* __restrict__ v1, float* __restrict__ h) {
  int lane = threadIdx.x & 31, wave = threadIdx.x >> 5;
  int tile = blockIdx.x * 8 + wave;
  int mt = tile / 12, dt = tile % 12;
  int row0 = mt * 16, col0 = dt * 16;
  v8f acc = {};
  #pragma unroll
  for (int kk = 0; kk < 6; ++kk) {
    v16bf a = load_frag_a(xT, row0, kk * 32, CC, lane);
    v16bf b = load_frag_b(w1, col0, kk * 32, CC, lane);
    acc = wmma_bf16(a, b, acc);
  }
  int d = col0 + (lane & 15);
  float sc  = g1[d] * rsqrtf(v1[d] + 1e-5f);
  float add = (fb[d] - m1[d]) * sc + b1[d];
  int mbase = row0 + ((lane & 16) ? 8 : 0);
  #pragma unroll
  for (int r = 0; r < 8; ++r)
    h[(size_t)(mbase + r) * CC + d] = acc[r] * sc + add;
}

// F[b,n,:] = [ h/||h|| , pe(n)/sqrt(C) ]  bf16 [M,384]; one wave per row
__global__ void normF_kernel(const float* __restrict__ h, bf16* __restrict__ F) {
  int lane = threadIdx.x & 31, wave = threadIdx.x >> 5;
  size_t row = (size_t)blockIdx.x * 8 + wave;
  const float* hr = h + row * CC;
  float s = 0.f;
  #pragma unroll
  for (int i = 0; i < 6; ++i) { float t = hr[lane + 32 * i]; s += t * t; }
  #pragma unroll
  for (int off = 16; off > 0; off >>= 1) s += __shfl_xor(s, off, 32);
  float inv = 1.0f / fmaxf(sqrtf(s), 1e-12f);
  bf16* Fr = F + row * C2C;
  int nsp = (int)(row % (size_t)NSP);
  float yy = (float)(nsp / 56), xx = (float)(nsp % 56);
  const float invSqrtC = 0.07216878364870323f;   // 1/sqrt(192)
  #pragma unroll
  for (int i = 0; i < 6; ++i) {
    int c = lane + 32 * i;
    Fr[c] = (bf16)(hr[c] * inv);
    int grp = c / 48, j = c % 48;
    float om  = __expf(-(float)j * (9.210340371976184f / 48.0f)); // 10000^(-j/48)
    float ang = ((grp < 2) ? yy : xx) * om;
    float pv  = ((grp & 1) == 0) ? __sinf(ang) : __cosf(ang);
    Fr[CC + c] = (bf16)(pv * invSqrtC);
  }
}

// Fused S = F F^T sweep + running top-9 per row.
// Block = 256 thr (8 waves). Each wave owns ONE 16-row tile (A frags resident in VGPRs);
// all waves share each 16x384 B column-panel staged in LDS (8x L2-traffic reduction).
// Grid: 25 blocks/batch (200 row-tile slots, 196 valid).
__global__ void dist_topk_kernel(const bf16* __restrict__ F, int* __restrict__ nn) {
  __shared__ __align__(16) bf16 panel[16][PLD];
  __shared__ float tileS[8][16][17];
  int lane = threadIdx.x & 31, wave = threadIdx.x >> 5;
  int b  = blockIdx.x / 25;
  int rt = (blockIdx.x % 25) * 8 + wave;
  bool valid = rt < 196;
  const bf16* Fb = F + (size_t)b * NSP * C2C;
  int row0 = rt * 16;

  v16bf afrag[12];
  if (valid) {
    #pragma unroll
    for (int kk = 0; kk < 12; ++kk) afrag[kk] = load_frag_a(Fb, row0, kk * 32, C2C, lane);
  }

  float val[KNN]; int idx[KNN];
  #pragma unroll
  for (int t = 0; t < KNN; ++t) { val[t] = -INFINITY; idx[t] = 0; }

  int myrow = lane & 15;
  int jbase = (lane & 16) ? 8 : 0;
  int nlane = lane & 15;
  int mo    = (lane & 16) ? 8 : 0;

  for (int ct = 0; ct < 196; ++ct) {
    __syncthreads();   // previous panel fully consumed
    // cooperative panel load: 16 rows x 384 bf16 = 768 chunks of 8 bf16 (16B), coalesced
    #pragma unroll
    for (int it = 0; it < 3; ++it) {
      int ch  = threadIdx.x + it * 256;
      int r   = ch / 48;
      int off = (ch % 48) * 8;
      *(v8bf*)&panel[r][off] = *(const v8bf*)(Fb + (size_t)(ct * 16 + r) * C2C + off);
    }
    __syncthreads();   // panel visible to all waves
    if (valid) {
      v8f acc = {};
      #pragma unroll
      for (int kk = 0; kk < 12; ++kk) {
        v16bf bfr = load_frag_b(&panel[0][0], 0, kk * 32, PLD, lane);
        acc = wmma_bf16(afrag[kk], bfr, acc);
      }
      // lane holds column nlane, rows mo..mo+7 -> transpose via per-wave LDS tile
      #pragma unroll
      for (int r = 0; r < 8; ++r) tileS[wave][mo + r][nlane] = acc[r];
      asm volatile("s_wait_dscnt 0" ::: "memory");
      #pragma unroll
      for (int j = 0; j < 8; ++j) {
        float v = tileS[wave][myrow][jbase + j];
        topk_insert(val, idx, v, ct * 16 + jbase + j);
      }
    }
  }

  // merge the two half-lists per row (lane l owns cols 0..7, lane l+16 owns 8..15)
  #pragma unroll
  for (int t = 0; t < KNN; ++t) {
    float ov = __shfl(val[t], (lane & 15) + 16, 32);
    int   oi = __shfl(idx[t], (lane & 15) + 16, 32);
    if (lane < 16) topk_insert(val, idx, ov, oi);
  }
  if (valid && lane < 16) {
    int* dst = nn + ((size_t)b * NSP + row0 + myrow) * KNN;
    #pragma unroll
    for (int t = 0; t < KNN; ++t) dst[t] = idx[t];
  }
}

// g = [h , max_k h[nn_k] - h]  bf16 [M,384]; block 192 thr, grid M
__global__ void aggregate_kernel(const float* __restrict__ h, const int* __restrict__ nn,
                                 bf16* __restrict__ g) {
  int row = blockIdx.x, c = threadIdx.x;
  int b = row / NSP;
  const int* ip = nn + (size_t)row * KNN;
  float hv = h[(size_t)row * CC + c];
  float mx = -INFINITY;
  #pragma unroll
  for (int k = 0; k < KNN; ++k)
    mx = fmaxf(mx, h[((size_t)b * NSP + ip[k]) * CC + c]);
  g[(size_t)row * C2C + c]      = (bf16)hv;
  g[(size_t)row * C2C + CC + c] = (bf16)(mx - hv);
}

// y = GELU(BNg(grouped(g) + gc_b))  bf16 [M,384]; one 16x16 tile / wave
__global__ void grouped_kernel(const bf16* __restrict__ g, const bf16* __restrict__ wg,
                               const float* __restrict__ gcb, const float* __restrict__ gg,
                               const float* __restrict__ gb, const float* __restrict__ gm,
                               const float* __restrict__ gv, bf16* __restrict__ y) {
  int lane = threadIdx.x & 31, wave = threadIdx.x >> 5;
  int tile = blockIdx.x * 8 + wave;
  int mt = tile / 24, rest = tile % 24, gi = rest / 6, dt = rest % 6;
  int row0 = mt * 16;
  const bf16* wgi = wg + gi * CG * CG;
  v8f acc = {};
  #pragma unroll
  for (int kk = 0; kk < 3; ++kk) {
    v16bf a = load_frag_a(g, row0, gi * CG + kk * 32, C2C, lane);
    v16bf b = load_frag_b(wgi, dt * 16, kk * 32, CG, lane);
    acc = wmma_bf16(a, b, acc);
  }
  int c2 = gi * CG + dt * 16 + (lane & 15);
  float sc  = gg[c2] * rsqrtf(gv[c2] + 1e-5f);
  float add = (gcb[c2] - gm[c2]) * sc + gb[c2];
  int mbase = row0 + ((lane & 16) ? 8 : 0);
  #pragma unroll
  for (int r = 0; r < 8; ++r) {
    float t  = acc[r] * sc + add;
    float ge = 0.5f * t * (1.0f + erff(t * 0.70710678118654752f));
    y[(size_t)(mbase + r) * C2C + c2] = (bf16)ge;
  }
}

// out = BN2(y @ fc2_w^T + fc2_b) + x ; fp32 (B,C,H,W); float4 transposed stores
__global__ void fc2_kernel(const bf16* __restrict__ y, const bf16* __restrict__ w2,
                           const float* __restrict__ fb, const float* __restrict__ g2,
                           const float* __restrict__ b2, const float* __restrict__ m2,
                           const float* __restrict__ v2, const float* __restrict__ x,
                           float* __restrict__ out) {
  int lane = threadIdx.x & 31, wave = threadIdx.x >> 5;
  int tile = blockIdx.x * 8 + wave;
  int mt = tile / 12, dt = tile % 12;
  int row0 = mt * 16, col0 = dt * 16;
  v8f acc = {};
  #pragma unroll
  for (int kk = 0; kk < 12; ++kk) {
    v16bf a = load_frag_a(y, row0, kk * 32, C2C, lane);
    v16bf b = load_frag_b(w2, col0, kk * 32, C2C, lane);
    acc = wmma_bf16(a, b, acc);
  }
  int d = col0 + (lane & 15);
  float sc  = g2[d] * rsqrtf(v2[d] + 1e-5f);
  float add = (fb[d] - m2[d]) * sc + b2[d];
  int gm0 = row0 + ((lane & 16) ? 8 : 0);
  int b   = gm0 / NSP;
  int nsp = gm0 - b * NSP;
  size_t base = ((size_t)b * CC + d) * NSP + nsp;     // 16B aligned (nsp % 8 == 0)
  float4 xa = *(const float4*)(x + base);
  float4 xb = *(const float4*)(x + base + 4);
  float4 o0 = { acc[0] * sc + add + xa.x, acc[1] * sc + add + xa.y,
                acc[2] * sc + add + xa.z, acc[3] * sc + add + xa.w };
  float4 o1 = { acc[4] * sc + add + xb.x, acc[5] * sc + add + xb.y,
                acc[6] * sc + add + xb.z, acc[7] * sc + add + xb.w };
  *(float4*)(out + base)     = o0;
  *(float4*)(out + base + 4) = o1;
}

// ---------------- launch ----------------

extern "C" void kernel_launch(void* const* d_in, const int* in_sizes, int n_in,
                              void* d_out, int out_size, void* d_ws, size_t ws_size,
                              hipStream_t stream) {
  const float* x     = (const float*)d_in[0];
  const float* fc1_w = (const float*)d_in[1];
  const float* fc1_b = (const float*)d_in[2];
  const float* bn1_g = (const float*)d_in[3];
  const float* bn1_b = (const float*)d_in[4];
  const float* bn1_m = (const float*)d_in[5];
  const float* bn1_v = (const float*)d_in[6];
  const float* gc_w  = (const float*)d_in[7];
  const float* gc_b  = (const float*)d_in[8];
  const float* bng_g = (const float*)d_in[9];
  const float* bng_b = (const float*)d_in[10];
  const float* bng_m = (const float*)d_in[11];
  const float* bng_v = (const float*)d_in[12];
  const float* fc2_w = (const float*)d_in[13];
  const float* fc2_b = (const float*)d_in[14];
  const float* bn2_g = (const float*)d_in[15];
  const float* bn2_b = (const float*)d_in[16];
  const float* bn2_m = (const float*)d_in[17];
  const float* bn2_v = (const float*)d_in[18];
  float* out = (float*)d_out;

  char* ws = (char*)d_ws;
  bf16* xT   = (bf16*)(ws);                        // 9,633,792 B
  bf16* w1b  = (bf16*)(ws +  9633792);             //    73,728 B
  bf16* wgb  = (bf16*)(ws +  9707520);             //    73,728 B
  bf16* w2b  = (bf16*)(ws +  9781248);             //   147,456 B
  float* h   = (float*)(ws + 9928704);             // 19,267,584 B
  bf16* F    = (bf16*)(ws + 29196288);             // 19,267,584 B (reused as y)
  int*  nn   = (int*) (ws + 48463872);             //   903,168 B
  bf16* g    = (bf16*)(ws + 49367040);             // 19,267,584 B  (end 68,634,624)
  bf16* yv   = F;                                  // reuse: F dead after top-k

  // 1) precision conversion / transpose
  f2bf_kernel<<<(CC * CC + 255) / 256, 256, 0, stream>>>(fc1_w, w1b, CC * CC);
  f2bf_kernel<<<(4 * CG * CG + 255) / 256, 256, 0, stream>>>(gc_w, wgb, 4 * CG * CG);
  f2bf_kernel<<<(CC * C2C + 255) / 256, 256, 0, stream>>>(fc2_w, w2b, CC * C2C);
  transpose_kernel<<<dim3(NSP / 32, CC / 32, BB), dim3(32, 8), 0, stream>>>(x, xT);

  // 2) fc1 + BN1
  fc1_kernel<<<(MROWS / 16) * 12 / 8, 256, 0, stream>>>(xT, w1b, fc1_b, bn1_g, bn1_b, bn1_m, bn1_v, h);

  // 3) row-normalize + pos-embed -> F
  normF_kernel<<<MROWS / 8, 256, 0, stream>>>(h, F);

  // 4) fused similarity GEMM + top-9 (LDS-shared B panels)
  dist_topk_kernel<<<BB * 25, 256, 0, stream>>>(F, nn);

  // 5) gather + max aggregate -> g
  aggregate_kernel<<<MROWS, CC, 0, stream>>>(h, nn, g);

  // 6) grouped conv + BNg + GELU -> y
  grouped_kernel<<<(MROWS / 16) * 24 / 8, 256, 0, stream>>>(g, wgb, gc_b, bng_g, bng_b, bng_m, bng_v, yv);

  // 7) fc2 + BN2 + residual -> out
  fc2_kernel<<<(MROWS / 16) * 12 / 8, 256, 0, stream>>>(yv, w2b, fc2_b, bn2_g, bn2_b, bn2_m, bn2_v, x, out);
}
